// Decoder_2611340116146
// MI455X (gfx1250) — matrix operand order repaired and checked by reading, not compiled, not run on hardware
//
#include <hip/hip_runtime.h>
#include <hip/hip_bf16.h>

typedef __bf16 bf16;
typedef __attribute__((ext_vector_type(16))) __bf16 v16bf;
typedef __attribute__((ext_vector_type(8)))  __bf16 v8bf;
typedef __attribute__((ext_vector_type(8)))  float  v8f;

// b128 async-copy pointer types (match the builtin's expected signature:
// 'vector_size(16) int' pointers; AS1 = global, AS3 = LDS)
typedef int v4i __attribute__((vector_size(16)));
typedef __attribute__((address_space(1))) v4i* gptr_b128;
typedef __attribute__((address_space(3))) v4i* lptr_b128;

#define BATCH 64
#define TSTEPS 20
#define HDIM 512
#define VOCAB 32000
#define FC_ROWS 32          // rows per block in fc GEMM (staged in LDS)
#define FC_NG (VOCAB / 64)  // 500 column groups of 64
#define FC_NOCT 63          // ceil(500 / 8) column octets per m-group

// ---------------- WMMA fragment helpers (wave32, 16x16x32 bf16) ----------------

// A fragment: row-major src[., ld], rows rbase..rbase+15, K window [k0, k0+32).
// Lanes 0-15 hold M=lane: elems 0..7 = K k0..k0+7, elems 8..15 = K k0+16..+23.
// Lanes 16-31: same rows, K offset +8 (per CDNA5 ISA 16-bit A-matrix table).
__device__ __forceinline__ v16bf load_frag_a(const bf16* __restrict__ src, int ld,
                                             int rbase, int k0) {
    int lane = threadIdx.x & 31;
    int r = lane & 15;
    int koff = (lane >> 4) << 3;   // 0 or 8
    const bf16* p = src + (size_t)(rbase + r) * ld + k0 + koff;
    union { v16bf v; v8bf h[2]; } u;
    u.h[0] = *(const v8bf*)(p);
    u.h[1] = *(const v8bf*)(p + 16);
    return u.v;
}

// Same fragment math, but source lives in LDS (compiler infers ds_load_b128).
__device__ __forceinline__ v16bf load_frag_a_lds(const bf16* src, int rbase, int k0) {
    int lane = threadIdx.x & 31;
    int r = lane & 15;
    int koff = (lane >> 4) << 3;
    const bf16* p = src + (rbase + r) * HDIM + k0 + koff;
    union { v16bf v; v8bf h[2]; } u;
    u.h[0] = *(const v8bf*)(p);
    u.h[1] = *(const v8bf*)(p + 16);
    return u.v;
}

// B fragment: B = W^T where W is [N, K] row-major. Lane holds column
// n = nbase + lane%16; elems 0..7 = K k0+koff..+7, elems 8..15 = K k0+koff+16..+23, koff = (lane>=16 ? 8 : 0).
__device__ __forceinline__ v16bf load_frag_b(const bf16* __restrict__ w, int ld,
                                             int nbase, int k0) {
    int lane = threadIdx.x & 31;
    int n = lane & 15;
    int koff = (lane >> 4) << 3;   // 0 or 8
    const bf16* p = w + (size_t)(nbase + n) * ld + k0 + koff;
    union { v16bf v; v8bf h[2]; } u;
    u.h[0] = *(const v8bf*)(p);
    u.h[1] = *(const v8bf*)(p + 16);
    return u.v;
}

__device__ __forceinline__ v8f wmma_bf16(v16bf a, v16bf b, v8f c) {
    return __builtin_amdgcn_wmma_f32_16x16x32_bf16(
        /*neg_a=*/false, a, /*neg_b=*/false, b,
        /*c_mod=*/(short)0, c, /*reuse_a=*/false, /*reuse_b=*/false);
}

__device__ __forceinline__ float sigmoidf_(float x) {
    return 1.0f / (1.0f + __expf(-x));
}

// ---------------- conversion / gather kernels ----------------

__global__ void cvt_bf16_kernel(const float* __restrict__ src, bf16* __restrict__ dst, int n) {
    int i = blockIdx.x * blockDim.x + threadIdx.x;
    if (i < n) dst[i] = (bf16)src[i];
}

__global__ void init_state_kernel(const float* __restrict__ x, bf16* __restrict__ xb,
                                  float* __restrict__ c, int n) {
    int i = blockIdx.x * blockDim.x + threadIdx.x;
    if (i < n) { xb[i] = (bf16)x[i]; c[i] = x[i]; }
}

// emb_bf16[(t*B + b)*E + e] = bf16(embed_w[captions[b*T + t]*E + e])
__global__ void embed_gather_kernel(const int* __restrict__ caps,
                                    const float* __restrict__ table,
                                    bf16* __restrict__ dst) {
    int i = blockIdx.x * blockDim.x + threadIdx.x;
    if (i >= TSTEPS * BATCH * HDIM) return;
    int r = i >> 9;            // row in [0, T*B)
    int e = i & (HDIM - 1);
    int t = r >> 6;            // r / 64
    int b = r & 63;
    int cap = caps[b * TSTEPS + t];
    dst[i] = (bf16)table[(size_t)cap * HDIM + e];
}

// ---------------- Gx GEMM: [1280,512] x [512,2048] + (b_ih + b_hh) -> f32 ----------------

__global__ void __launch_bounds__(256)
gx_gemm_kernel(const bf16* __restrict__ A, const bf16* __restrict__ W,
               const float* __restrict__ b0, const float* __restrict__ b1,
               float* __restrict__ out, int M, int N) {
    int wid = (blockIdx.x * blockDim.x + threadIdx.x) >> 5;
    int ngroups = N >> 6;                    // 64 cols per wave (4 tiles)
    int mt = wid / ngroups;
    int ng = wid % ngroups;
    if (mt * 16 >= M) return;
    int lane = threadIdx.x & 31;
    int col = lane & 15;
    int rowadd = (lane >> 4) << 3;
    int mbase = mt * 16;
    int nbase = ng * 64;

    v8f acc[4];
#pragma unroll
    for (int q = 0; q < 4; ++q) {
        float bv = b0[nbase + q * 16 + col] + b1[nbase + q * 16 + col];
#pragma unroll
        for (int g = 0; g < 8; ++g) acc[q][g] = bv;
    }
    for (int k0 = 0; k0 < HDIM; k0 += 32) {
        v16bf a = load_frag_a(A, HDIM, mbase, k0);
#pragma unroll
        for (int q = 0; q < 4; ++q) {
            v16bf b = load_frag_b(W, HDIM, nbase + q * 16, k0);
            acc[q] = wmma_bf16(a, b, acc[q]);
        }
    }
#pragma unroll
    for (int q = 0; q < 4; ++q)
#pragma unroll
        for (int g = 0; g < 8; ++g) {
            int r = mbase + rowadd + g;
            out[(size_t)r * N + nbase + q * 16 + col] = acc[q][g];
        }
}

// ---------------- LSTM step: gates = Gx[t] + h@W_hh^T ; update c, h ----------------
// Each wave owns a 16x16 (batch x hidden) patch and computes all 4 gate tiles
// (i/f/g/o live at n, n+512, n+1024, n+1536) so it can update c/h locally.
// 128 waves total: mt in [0,4), nt in [0,32).

__global__ void __launch_bounds__(128)
lstm_step_kernel(const bf16* __restrict__ Hprev, const bf16* __restrict__ Whh,
                 const float* __restrict__ Gx, float* __restrict__ C,
                 bf16* __restrict__ Hout) {
    int wid = (blockIdx.x * blockDim.x + threadIdx.x) >> 5;  // 0..127
    int mt = wid >> 5;
    int nt = wid & 31;
    int lane = threadIdx.x & 31;
    int col = lane & 15;
    int rowadd = (lane >> 4) << 3;
    int mbase = mt * 16;
    int nbase = nt * 16;

    v8f acc[4];
#pragma unroll
    for (int q = 0; q < 4; ++q)
#pragma unroll
        for (int g = 0; g < 8; ++g)
            acc[q][g] = Gx[(size_t)(mbase + rowadd + g) * (4 * HDIM) + q * HDIM + nbase + col];

    for (int k0 = 0; k0 < HDIM; k0 += 32) {
        v16bf a = load_frag_a(Hprev, HDIM, mbase, k0);
#pragma unroll
        for (int q = 0; q < 4; ++q) {
            v16bf b = load_frag_b(Whh, HDIM, q * HDIM + nbase, k0);
            acc[q] = wmma_bf16(a, b, acc[q]);
        }
    }

#pragma unroll
    for (int g = 0; g < 8; ++g) {
        int bidx = mbase + rowadd + g;
        int idx = bidx * HDIM + nbase + col;
        float iv = sigmoidf_(acc[0][g]);
        float fv = sigmoidf_(acc[1][g]);
        float gv = tanhf(acc[2][g]);
        float ov = sigmoidf_(acc[3][g]);
        float cn = fv * C[idx] + iv * gv;
        C[idx] = cn;
        Hout[idx] = (bf16)(ov * tanhf(cn));
    }
}

// ---------------- FC GEMM: [1280,512] x [512,32000] + fc_b -> out[b,t,v] ----------------
// Block = 8 waves sharing one 32-row A tile (32 KB) staged into LDS with the
// CDNA5 async-copy path; each wave computes a 32x64 output strip (8 WMMA/k-step,
// 2x B reuse vs a 16-row wave).

__global__ void __launch_bounds__(256)
fc_gemm_kernel(const bf16* __restrict__ Hseq, const bf16* __restrict__ Wfc,
               const float* __restrict__ bias, float* __restrict__ out) {
    __shared__ bf16 As[FC_ROWS * HDIM];          // 32 KB

    int mtg = blockIdx.x / FC_NOCT;              // 0..39
    int oct = blockIdx.x % FC_NOCT;              // 0..62
    int wave = threadIdx.x >> 5;                 // 0..7
    int ng = oct * 8 + wave;                     // 64-col group id (may be >= 500)
    int mbase = mtg * FC_ROWS;

    // ---- cooperative stage of A tile: rows mbase..mbase+31 are contiguous ----
    {
        const bf16* gsrc = Hseq + (size_t)mbase * HDIM;
#pragma unroll
        for (int it = 0; it < (FC_ROWS * HDIM / 8) / 256; ++it) {
            int j = it * 256 + threadIdx.x;      // 16-byte chunk index
#if __has_builtin(__builtin_amdgcn_global_load_async_to_lds_b128)
            __builtin_amdgcn_global_load_async_to_lds_b128(
                (gptr_b128)(gsrc + j * 8),
                (lptr_b128)(As + j * 8), 0, 0);
#else
            *(v8bf*)(As + j * 8) = *(const v8bf*)(gsrc + j * 8);
#endif
        }
#if __has_builtin(__builtin_amdgcn_global_load_async_to_lds_b128)
#if __has_builtin(__builtin_amdgcn_s_wait_asynccnt)
        __builtin_amdgcn_s_wait_asynccnt(0);
#else
        asm volatile("s_wait_asynccnt 0" ::: "memory");
#endif
#endif
        __syncthreads();
    }

    if (ng < FC_NG) {
        int lane = threadIdx.x & 31;
        int col = lane & 15;
        int rowadd = (lane >> 4) << 3;
        int nbase = ng * 64;

        v8f acc[2][4];
#pragma unroll
        for (int q = 0; q < 4; ++q) {
            float bv = bias[nbase + q * 16 + col];
#pragma unroll
            for (int g = 0; g < 8; ++g) { acc[0][q][g] = bv; acc[1][q][g] = bv; }
        }

        for (int k0 = 0; k0 < HDIM; k0 += 32) {
            v16bf a0 = load_frag_a_lds(As, 0, k0);
            v16bf a1 = load_frag_a_lds(As, 16, k0);
#pragma unroll
            for (int q = 0; q < 4; ++q) {
                v16bf b = load_frag_b(Wfc, HDIM, nbase + q * 16, k0);
                acc[0][q] = wmma_bf16(a0, b, acc[0][q]);
                acc[1][q] = wmma_bf16(a1, b, acc[1][q]);
            }
        }

#pragma unroll
        for (int m = 0; m < 2; ++m)
#pragma unroll
            for (int q = 0; q < 4; ++q)
#pragma unroll
                for (int g = 0; g < 8; ++g) {
                    int r = mbase + m * 16 + rowadd + g;   // r = t*64 + b
                    int t = r >> 6;
                    int b = r & 63;
                    int v = nbase + q * 16 + col;
                    out[(size_t)b * (TSTEPS * VOCAB) + (size_t)t * VOCAB + v] =
                        acc[m][q][g];
                }
    }
}

// ---------------- host side ----------------

extern "C" void kernel_launch(void* const* d_in, const int* in_sizes, int n_in,
                              void* d_out, int out_size, void* d_ws, size_t ws_size,
                              hipStream_t stream) {
    (void)in_sizes; (void)n_in; (void)out_size; (void)ws_size;
    const float* x        = (const float*)d_in[0];
    const int*   captions = (const int*)d_in[1];
    const float* embed_w  = (const float*)d_in[2];
    const float* W_ih     = (const float*)d_in[3];
    const float* W_hh     = (const float*)d_in[4];
    const float* b_ih     = (const float*)d_in[5];
    const float* b_hh     = (const float*)d_in[6];
    const float* fc_w     = (const float*)d_in[7];
    const float* fc_b     = (const float*)d_in[8];
    float* out = (float*)d_out;

    // workspace carve-up (256B aligned segments)
    char* ws = (char*)d_ws;
    size_t off = 0;
    auto carve = [&](size_t bytes) -> void* {
        void* p = ws + off;
        off += (bytes + 255) & ~(size_t)255;
        return p;
    };
    bf16*  fcw_b  = (bf16*)carve((size_t)VOCAB * HDIM * 2);            // 32.77 MB
    bf16*  wih_b  = (bf16*)carve((size_t)4 * HDIM * HDIM * 2);         //  2.10 MB
    bf16*  whh_b  = (bf16*)carve((size_t)4 * HDIM * HDIM * 2);         //  2.10 MB
    bf16*  x_b    = (bf16*)carve((size_t)BATCH * HDIM * 2);
    bf16*  emb_b  = (bf16*)carve((size_t)TSTEPS * BATCH * HDIM * 2);   //  1.31 MB
    bf16*  hseq_b = (bf16*)carve((size_t)TSTEPS * BATCH * HDIM * 2);   //  1.31 MB
    float* c_f32  = (float*)carve((size_t)BATCH * HDIM * 4);
    float* gx_f32 = (float*)carve((size_t)TSTEPS * BATCH * 4 * HDIM * 4); // 10.49 MB

    const int nFc  = VOCAB * HDIM;          // 16,384,000
    const int nW   = 4 * HDIM * HDIM;       // 1,048,576
    const int nX   = BATCH * HDIM;          // 32,768
    const int nEmb = TSTEPS * BATCH * HDIM; // 655,360

    cvt_bf16_kernel<<<(nFc + 255) / 256, 256, 0, stream>>>(fc_w, fcw_b, nFc);
    cvt_bf16_kernel<<<(nW + 255) / 256, 256, 0, stream>>>(W_ih, wih_b, nW);
    cvt_bf16_kernel<<<(nW + 255) / 256, 256, 0, stream>>>(W_hh, whh_b, nW);
    init_state_kernel<<<(nX + 255) / 256, 256, 0, stream>>>(x, x_b, c_f32, nX);
    embed_gather_kernel<<<(nEmb + 255) / 256, 256, 0, stream>>>(captions, embed_w, emb_b);

    // Gx = emb @ W_ih^T + b_ih + b_hh : M=1280, N=2048 -> 80*32 = 2560 waves
    gx_gemm_kernel<<<320, 256, 0, stream>>>(emb_b, wih_b, b_ih, b_hh, gx_f32,
                                            TSTEPS * BATCH, 4 * HDIM);

    // sequential recurrence: 20 small launches, stream order gives the dependency
    for (int t = 0; t < TSTEPS; ++t) {
        const bf16* hp = (t == 0) ? x_b : hseq_b + (size_t)(t - 1) * BATCH * HDIM;
        lstm_step_kernel<<<32, 128, 0, stream>>>(
            hp, whh_b, gx_f32 + (size_t)t * BATCH * 4 * HDIM, c_f32,
            hseq_b + (size_t)t * BATCH * HDIM);
    }

    // logits = Hseq @ fc_w^T + fc_b : 40 m-groups x 63 column-octets
    fc_gemm_kernel<<<40 * FC_NOCT, 256, 0, stream>>>(hseq_b, fcw_b, fc_b, out);
}
